// GIN_88098369176167
// MI455X (gfx1250) — matrix operand order repaired and checked by reading, not compiled, not run on hardware
//
#include <hip/hip_runtime.h>
#include <hip/hip_bf16.h>

#define NNODES 50000
#define NEDGES 800000
#define NGRAPH 256
#define HID    128
#define BN_EPS 1e-5f
#define WELEM  (HID * HID)

typedef __bf16 bf16;
typedef __attribute__((ext_vector_type(16))) __bf16 v16bf;
typedef __attribute__((ext_vector_type(2)))  __bf16 bf16x2;
typedef __attribute__((ext_vector_type(8)))  float  v8f;

// ---------------- generic zero fill ----------------
__global__ void zero_kernel(float* __restrict__ p, long n) {
    long i = (long)blockIdx.x * blockDim.x + threadIdx.x;
    long stride = (long)gridDim.x * blockDim.x;
    for (; i < n; i += stride) p[i] = 0.0f;
}

// ---------------- weight pre-pass: fp32 [K][N] -> bf16 transposed [N][K] ----------------
__global__ void convert_w_kernel(const float* __restrict__ W, bf16* __restrict__ Wt) {
    int i = blockIdx.x * blockDim.x + threadIdx.x;
    if (i < WELEM) {
        int k = i >> 7, n = i & (HID - 1);
        Wt[n * HID + k] = (bf16)W[i];
    }
}

// ---------------- edge scatter-add: agg[dst] += h[src] ----------------
__global__ void aggregate_kernel(const float* __restrict__ h,
                                 const int* __restrict__ src,
                                 const int* __restrict__ dst,
                                 float* __restrict__ agg, int nedges) {
    long total = (long)nedges * (HID / 4);
    long stride = (long)gridDim.x * blockDim.x;
    for (long i = (long)blockIdx.x * blockDim.x + threadIdx.x; i < total; i += stride) {
        int e  = (int)(i >> 5);
        int c4 = ((int)(i & 31)) << 2;
        int s = src[e];
        int d = dst[e];
        const float4 v = *(const float4*)(h + (long)s * HID + c4);
        float* ap = agg + (long)d * HID + c4;
        atomicAdd(ap + 0, v.x);
        atomicAdd(ap + 1, v.y);
        atomicAdd(ap + 2, v.z);
        atomicAdd(ap + 3, v.w);
    }
}

// ---------------- x = (1+eps)*h + agg  (in place on agg) ----------------
__global__ void combine_kernel(const float* __restrict__ h,
                               float* __restrict__ agg,
                               const float* __restrict__ eps_ptr, long n) {
    float c = 1.0f + eps_ptr[0];
    long stride = (long)gridDim.x * blockDim.x;
    for (long i = (long)blockIdx.x * blockDim.x + threadIdx.x; i < n; i += stride)
        agg[i] = c * h[i] + agg[i];
}

// ---------------- WMMA bf16 GEMM ----------------
// Y[:, :128](ldY) = X[:, :128](ldX) @ Wt^T [+bias][+=Y][relu]
// Wt is bf16, pre-transposed ([N][K]). Block = 256 threads = 8 waves; each wave
// owns one 16-col tile. Block is persistent over `strips` consecutive 16-row
// strips, reusing the LDS-staged weight (weight traffic amortized 8x).
__global__ __launch_bounds__(256) void gemm128_wmma_kernel(
    const float* __restrict__ X, int ldX,
    const bf16*  __restrict__ Wt,         // [128][128] bf16 transposed
    const float* __restrict__ bias,       // may be null
    float* __restrict__ Y, int ldY,
    int nrows, int relu, int addto, int strips) {
    __shared__ __align__(16) bf16 sX[16 * HID];     // A strip, row-major [16][128]
    __shared__ __align__(16) bf16 sWt[HID * HID];   // weight, [n][k]

    const int tid = threadIdx.x;

    // stage pre-converted bf16 weight with 16B vector copies
    {
        const uint4* wsrc = (const uint4*)Wt;
        uint4* wdst = (uint4*)sWt;
        for (int i = tid; i < WELEM / 8; i += 256) wdst[i] = wsrc[i];
    }

    const int lane = tid & 31;
    const int wave = tid >> 5;
    const int col0 = wave << 4;
    const int m    = lane & 15;
    const int khA  = (lane < 16) ? 0 : 8;    // A: lane halves split K 0-7 / 8-15
    const int khB  = (lane < 16) ? 0 : 16;   // unused: B loads use khA, same K split as A
    const int ncol = col0 + m;
    const int tile0 = blockIdx.x * strips;

    union Frag { v16bf v; bf16x2 p[8]; };

    for (int s = 0; s < strips; ++s) {
        const int row0 = (tile0 + s) << 4;
        if (row0 >= nrows) break;
        __syncthreads();   // previous strip's readers done (also orders W stage)
        // stage activation strip as bf16
        for (int i = tid; i < 16 * HID; i += 256) {
            int r = i >> 7, c = i & (HID - 1);
            int gr = row0 + r;
            sX[i] = (bf16)(gr < nrows ? X[(long)gr * ldX + c] : 0.0f);
        }
        __syncthreads();

        // prefetch next strip's activation rows (global_prefetch_b8)
        if (tid < 32 && row0 + 16 < nrows) {
            int pr = row0 + 16 + (tid >> 1);
            if (pr < nrows)
                __builtin_prefetch(X + (long)pr * ldX + (tid & 1) * 64, 0, 1);
        }

        v8f acc = {};
#pragma unroll
        for (int kb = 0; kb < HID; kb += 32) {
            Frag a, b;
#pragma unroll
            for (int v = 0; v < 8; ++v) {
                // A 16x32 bf16: VGPR v<4 -> K=khA+2v ; v>=4 -> K=khA+16+2(v-4)
                int kkA = kb + khA + ((v < 4) ? (2 * v) : (8 + 2 * v));
                a.p[v] = *(const bf16x2*)&sX[m * HID + kkA];
                // B 32x16 bf16: lane col = ncol, same K mapping as A
                int kkB = kb + khA + ((v < 4) ? (2 * v) : (8 + 2 * v));
                b.p[v] = *(const bf16x2*)&sWt[ncol * HID + kkB];
            }
            acc = __builtin_amdgcn_wmma_f32_16x16x32_bf16(
                false, a.v, false, b.v, (short)0, acc, false, false);
        }

        // D 16x16 f32: lane col = col0+(lane&15); VGPR v -> row v (+8 upper lanes)
        const int outcol = col0 + m;
        const int rbase  = row0 + ((lane < 16) ? 0 : 8);
        const float bv   = bias ? bias[outcol] : 0.0f;
#pragma unroll
        for (int v = 0; v < 8; ++v) {
            int gr = rbase + v;
            if (gr < nrows) {
                long idx = (long)gr * ldY + outcol;
                float val = acc[v] + bv;
                if (addto) val += Y[idx];
                if (relu)  val = fmaxf(val, 0.0f);
                Y[idx] = val;
            }
        }
    }
}

// ---------------- BatchNorm stats: per-column sum / sumsq ----------------
__global__ void bnstats_kernel(const float* __restrict__ X,
                               float* __restrict__ sums,
                               float* __restrict__ sumsq, int nrows) {
    int c  = threadIdx.x;            // 128 threads = one per column
    int r0 = blockIdx.x * 128;
    int r1 = r0 + 128; if (r1 > nrows) r1 = nrows;
    float s = 0.0f, q = 0.0f;
    for (int r = r0; r < r1; ++r) {
        float v = X[(long)r * HID + c];
        s += v; q += v * v;
    }
    atomicAdd(&sums[c], s);
    atomicAdd(&sumsq[c], q);
}

// ---------------- BN normalize (training stats) + ReLU, in place ----------------
__global__ void bnnorm_relu_kernel(float* __restrict__ X,
                                   const float* __restrict__ sums,
                                   const float* __restrict__ sumsq,
                                   const float* __restrict__ gamma,
                                   const float* __restrict__ beta,
                                   long n, float invN) {
    long stride = (long)gridDim.x * blockDim.x;
    for (long i = (long)blockIdx.x * blockDim.x + threadIdx.x; i < n; i += stride) {
        int c = (int)(i & (HID - 1));
        float mu  = sums[c] * invN;
        float var = sumsq[c] * invN - mu * mu;
        float v = (X[i] - mu) * rsqrtf(var + BN_EPS) * gamma[c] + beta[c];
        X[i] = fmaxf(v, 0.0f);
    }
}

// ---------------- per-graph readout: hg[gid[n]] += hcat[n]  ([N,256] -> [G,256]) ----
__global__ void readout_kernel(const float* __restrict__ hcat,
                               const int* __restrict__ gids,
                               float* __restrict__ hg, int nnodes) {
    long total = (long)nnodes * 256;
    long stride = (long)gridDim.x * blockDim.x;
    for (long i = (long)blockIdx.x * blockDim.x + threadIdx.x; i < total; i += stride) {
        int n = (int)(i >> 8);
        int c = (int)(i & 255);
        atomicAdd(&hg[(long)gids[n] * 256 + c], hcat[i]);
    }
}

// ---------------- final 128 -> 1 projection ----------------
__global__ void final_kernel(const float* __restrict__ m2,
                             const float* __restrict__ w3,
                             const float* __restrict__ b3,
                             float* __restrict__ out, int ngraphs) {
    int g = blockIdx.x * blockDim.x + threadIdx.x;
    if (g < ngraphs) {
        float s = b3[0];
#pragma unroll 4
        for (int k = 0; k < HID; ++k) s += m2[(long)g * HID + k] * w3[k];
        out[g] = s;
    }
}

extern "C" void kernel_launch(void* const* d_in, const int* in_sizes, int n_in,
                              void* d_out, int out_size, void* d_ws, size_t ws_size,
                              hipStream_t stream) {
    const float* feats = (const float*)d_in[0];
    const int*   src   = (const int*)d_in[1];   // [2, E]
    const int*   dst   = (const int*)d_in[2];   // [2, E]
    const int*   gids  = (const int*)d_in[3];   // [N]
    const float* eps   = (const float*)d_in[5]; // [2,2]
    const float* Wa    = (const float*)d_in[6]; // [2,2,128,128]
    const float* ba    = (const float*)d_in[7]; // [2,2,128]
    const float* bng   = (const float*)d_in[8];
    const float* bnb   = (const float*)d_in[9];
    const float* Wb    = (const float*)d_in[10];
    const float* bb    = (const float*)d_in[11];
    const float* oW1   = (const float*)d_in[12]; // [256,128]
    const float* ob1   = (const float*)d_in[13];
    const float* oW2   = (const float*)d_in[14]; // [128,128]
    const float* ob2   = (const float*)d_in[15];
    const float* oW3   = (const float*)d_in[16]; // [128,1]
    const float* ob3   = (const float*)d_in[17];
    float* out = (float*)d_out;

    // ---- workspace layout ----
    float* ws    = (float*)d_ws;
    float* buf0  = ws;                                 // h      [N,128]
    float* buf1  = buf0 + (long)NNODES * HID;          // agg/x  [N,128]
    float* buf2  = buf1 + (long)NNODES * HID;          // linA   [N,128]
    float* hcat  = buf2 + (long)NNODES * HID;          // concat [N,256]
    float* hg    = hcat + (long)NNODES * 256;          // [G,256]
    float* m1    = hg   + (long)NGRAPH * 256;          // [G,128]
    float* m2    = m1   + (long)NGRAPH * HID;          // [G,128]
    float* sums  = m2   + (long)NGRAPH * HID;          // [128]
    float* sumsq = sums + HID;                         // [128]
    bf16*  wT    = (bf16*)(sumsq + HID);               // 11 x [128][128] bf16
    bf16*  WaT   = wT;                                 // [4] per (e,l)
    bf16*  WbT   = wT + (long)4  * WELEM;              // [4] per (e,l)
    bf16*  oW1T0 = wT + (long)8  * WELEM;              // oW1 rows 0..127
    bf16*  oW1T1 = wT + (long)9  * WELEM;              // oW1 rows 128..255
    bf16*  oW2T  = wT + (long)10 * WELEM;

    const long nodeElems  = (long)NNODES * HID;
    const int  STRIPS     = 8;
    const int  tiles      = NNODES / 16;                       // 3125 exact
    const int  gemmBlocks = (tiles + STRIPS - 1) / STRIPS;     // 391
    const int  bnBlocks   = (NNODES + 127) / 128;
    const int  cvtBlocks  = WELEM / 256;
    const float invN = 1.0f / (float)NNODES;

    // ---- pre-convert all weights to bf16 transposed (once per launch) ----
    for (int el = 0; el < 4; ++el) {
        convert_w_kernel<<<cvtBlocks, 256, 0, stream>>>(Wa + (long)el * WELEM, WaT + (long)el * WELEM);
        convert_w_kernel<<<cvtBlocks, 256, 0, stream>>>(Wb + (long)el * WELEM, WbT + (long)el * WELEM);
    }
    convert_w_kernel<<<cvtBlocks, 256, 0, stream>>>(oW1, oW1T0);
    convert_w_kernel<<<cvtBlocks, 256, 0, stream>>>(oW1 + (long)WELEM, oW1T1);
    convert_w_kernel<<<cvtBlocks, 256, 0, stream>>>(oW2, oW2T);

    for (int e = 0; e < 2; ++e) {
        const float* h = feats;                        // layer input
        for (int l = 0; l < 2; ++l) {
            const int el = e * 2 + l;
            // agg = segment_sum(h[src], dst)
            zero_kernel<<<4096, 256, 0, stream>>>(buf1, nodeElems);
            aggregate_kernel<<<8192, 256, 0, stream>>>(h, src + (long)e * NEDGES,
                                                       dst + (long)e * NEDGES, buf1, NEDGES);
            // x = (1+eps)*h + agg   (in place on buf1)
            combine_kernel<<<4096, 256, 0, stream>>>(h, buf1, eps + el, nodeElems);
            // linear A + bias
            gemm128_wmma_kernel<<<gemmBlocks, 256, 0, stream>>>(
                buf1, HID, WaT + (long)el * WELEM, ba + (long)el * HID,
                buf2, HID, NNODES, /*relu=*/0, /*addto=*/0, STRIPS);
            // batchnorm (training stats) + relu
            zero_kernel<<<1, 256, 0, stream>>>(sums, 2 * HID);
            bnstats_kernel<<<bnBlocks, 128, 0, stream>>>(buf2, sums, sumsq, NNODES);
            bnnorm_relu_kernel<<<4096, 256, 0, stream>>>(buf2, sums, sumsq,
                                                         bng + (long)el * HID,
                                                         bnb + (long)el * HID,
                                                         nodeElems, invN);
            // linear B + bias (+relu only for layer 0); last layer writes concat slab
            if (l == 0) {
                gemm128_wmma_kernel<<<gemmBlocks, 256, 0, stream>>>(
                    buf2, HID, WbT + (long)el * WELEM, bb + (long)el * HID,
                    buf0, HID, NNODES, /*relu=*/1, /*addto=*/0, STRIPS);
                h = buf0;
            } else {
                gemm128_wmma_kernel<<<gemmBlocks, 256, 0, stream>>>(
                    buf2, HID, WbT + (long)el * WELEM, bb + (long)el * HID,
                    hcat + (long)e * HID, 256, NNODES, /*relu=*/0, /*addto=*/0, STRIPS);
            }
        }
    }

    // per-graph sum readout
    zero_kernel<<<256, 256, 0, stream>>>(hg, (long)NGRAPH * 256);
    readout_kernel<<<8192, 256, 0, stream>>>(hcat, gids, hg, NNODES);

    // out MLP: relu(hg @ oW1 + ob1) -> m1   (K=256 split into two K=128 WMMA passes)
    gemm128_wmma_kernel<<<NGRAPH / 16, 256, 0, stream>>>(
        hg, 256, oW1T0, (const float*)nullptr, m1, HID, NGRAPH, 0, 0, 1);
    gemm128_wmma_kernel<<<NGRAPH / 16, 256, 0, stream>>>(
        hg + HID, 256, oW1T1, ob1, m1, HID, NGRAPH, /*relu=*/1, /*addto=*/1, 1);
    // relu(m1 @ oW2 + ob2) -> m2
    gemm128_wmma_kernel<<<NGRAPH / 16, 256, 0, stream>>>(
        m1, HID, oW2T, ob2, m2, HID, NGRAPH, /*relu=*/1, /*addto=*/0, 1);
    // m2 @ oW3 + ob3 -> out [G,1]
    final_kernel<<<1, 256, 0, stream>>>(m2, oW3, ob3, out, NGRAPH);
}
